// AttentionLayer_21242908246135
// MI455X (gfx1250) — matrix operand order repaired and checked by reading, not compiled, not run on hardware
//
#include <hip/hip_runtime.h>
#include <hip/hip_bf16.h>

// ---------------------------------------------------------------------------
// Problem constants (AttentionLayer: B=2, L=2048, D=1024, H=16, Dh=64)
// ---------------------------------------------------------------------------
static constexpr int BSZ = 2;
static constexpr int H   = 16;
static constexpr int L   = 2048;
static constexpr int Dh  = 64;
static constexpr int D   = 1024;

typedef __attribute__((ext_vector_type(16))) __bf16 v16bf;
typedef __attribute__((ext_vector_type(8)))  float  v8f;
typedef int v4i __attribute__((vector_size(16)));

union FragBF {
  v16bf v;
  uint4 q[2];
};

#define DEV __device__ __forceinline__

// ---------------------------------------------------------------------------
// gfx1250 async global->LDS copy (ASYNCcnt-tracked).
// Builtin signature (from hipcc diagnostic): (v4i __device__*, v4i __shared__*,
// imm offset, imm cpol).
// ---------------------------------------------------------------------------
#if __has_builtin(__builtin_amdgcn_global_load_async_to_lds_b128)
#define ASYNC_USE_BUILTIN 1
#endif

DEV void async_copy16(const void* gsrc, void* ldst) {
#ifdef ASYNC_USE_BUILTIN
  __builtin_amdgcn_global_load_async_to_lds_b128(
      (__attribute__((address_space(1))) v4i*)gsrc,
      (__attribute__((address_space(3))) v4i*)ldst, 0, 0);
#else
  unsigned lds_off =
      (unsigned)(uintptr_t)(__attribute__((address_space(3))) void*)ldst;
  unsigned long long ga = (unsigned long long)(uintptr_t)gsrc;
  asm volatile("global_load_async_to_lds_b128 %0, %1, off"
               :: "v"(lds_off), "v"(ga)
               : "memory");
#endif
}

DEV void wait_async0() {
#if __has_builtin(__builtin_amdgcn_s_wait_asynccnt)
  __builtin_amdgcn_s_wait_asynccnt(0);
#else
  asm volatile("s_wait_asynccnt 0" ::: "memory");
#endif
}

// D = A(16x32 bf16) * B(32x16 bf16) + C(16x16 f32)
DEV v8f wmma_bf16(v16bf a, v16bf b, v8f c) {
  return __builtin_amdgcn_wmma_f32_16x16x32_bf16(
      /*neg_a=*/false, a, /*neg_b=*/false, b,
      /*c_mod=*/(short)0, c, /*reuse_a=*/false, /*reuse_b=*/false);
}

// A fragment (16x32, 16-bit): lanes 0-15 hold row m=lane, K 0-7 & 16-23;
// lanes 16-31 same rows, K 8-15 & 24-31.  `tile` = row-major [16][stride].
DEV v16bf load_frag_a(const __bf16* tile, int stride) {
  int lane = threadIdx.x & 31;
  int m    = lane & 15;
  int kb   = (lane >> 4) << 3;  // 0 or 8
  const __bf16* p = tile + (size_t)m * stride + kb;
  FragBF f;
  f.q[0] = *reinterpret_cast<const uint4*>(p);       // K = kb .. kb+7
  f.q[1] = *reinterpret_cast<const uint4*>(p + 16);  // K = 16+kb .. 16+kb+7
  return f.v;
}

// B fragment (32x16, 16-bit) from a COLUMN-major (N-major) tile:
// tileT[n][k].  Lane holds column n = lane&15, K = kb..kb+7 & 16+kb..16+kb+7, kb = 8*(lane>>4).
DEV v16bf load_frag_b(const __bf16* tileT, int stride) {
  int lane = threadIdx.x & 31;
  int n    = lane & 15;
  int kb   = (lane >> 4) << 3;  // 0 or 8
  const __bf16* p = tileT + (size_t)n * stride + kb;
  FragBF f;
  f.q[0] = *reinterpret_cast<const uint4*>(p);
  f.q[1] = *reinterpret_cast<const uint4*>(p + 16);
  return f.v;
}

// A fragment built from f32 memory (global or LDS), converted to bf16.
DEV v16bf load_frag_a_f32(const float* base, int stride) {
  int lane = threadIdx.x & 31;
  int m    = lane & 15;
  int kb   = (lane >> 4) << 3;
  const float* p = base + (size_t)m * stride + kb;
  float4 x0 = *reinterpret_cast<const float4*>(p);
  float4 x1 = *reinterpret_cast<const float4*>(p + 4);
  float4 x2 = *reinterpret_cast<const float4*>(p + 16);
  float4 x3 = *reinterpret_cast<const float4*>(p + 20);
  float t[16] = {x0.x, x0.y, x0.z, x0.w, x1.x, x1.y, x1.z, x1.w,
                 x2.x, x2.y, x2.z, x2.w, x3.x, x3.y, x3.z, x3.w};
  v16bf f;
#pragma unroll
  for (int i = 0; i < 16; ++i) f[i] = (__bf16)t[i];
  return f;
}

// ---------------------------------------------------------------------------
// Kernel 1/5: C = A(4096x1024) @ W(1024x1024) + bias.
// MODE 0: bf16 out, head-split  [bh][l][64]
// MODE 1: bf16 out, V-transpose [bh][dh][L]
// MODE 2: f32 out,  plain       [m][1024]
// Workgroup: 256 threads (8 waves), tile M=128 x N=64, K-step 32.
// A-tile staged RAW via async global->LDS DMA (16B per lane per op);
// B-tile staged with manual f32->bf16 transpose.
// ---------------------------------------------------------------------------
template <typename AT, int MODE>
__global__ __launch_bounds__(256)
void gemm_bias_kernel(const AT* __restrict__ A, const float* __restrict__ Wt,
                      const float* __restrict__ bias,
                      __bf16* __restrict__ outbf, float* __restrict__ outf) {
  constexpr int Ksz = 1024, Nsz = 1024;
  __shared__ __align__(16) AT     As[128][32];   // raw A tile (f32 or bf16)
  __shared__ __align__(16) __bf16 Bts[64][32];   // transposed B tile [n][k]

  const int tid  = threadIdx.x;
  const int wave = tid >> 5;
  const int m0   = blockIdx.y * 128;
  const int n0   = blockIdx.x * 64;

  v8f zero = {};
  v8f acc[4];
#pragma unroll
  for (int i = 0; i < 4; ++i) acc[i] = zero;

  // Async A staging geometry: one 16-byte chunk per lane per op.
  constexpr int CHUNKS     = (128 * 32 * (int)sizeof(AT)) / 16;  // total 16B chunks
  constexpr int NPASS      = CHUNKS / 256;                       // 4 (f32) / 2 (bf16)
  constexpr int CH_PER_ROW = (32 * (int)sizeof(AT)) / 16;        // 8 (f32) / 4 (bf16)
  constexpr int EL_PER_CH  = 16 / (int)sizeof(AT);               // 4 (f32) / 8 (bf16)

  for (int kt = 0; kt < Ksz / 32; ++kt) {
    const int k0 = kt * 32;

    // ---- stage A tile via async DMA (raw element type) ----
#pragma unroll
    for (int pass = 0; pass < NPASS; ++pass) {
      int c   = pass * 256 + tid;
      int row = c / CH_PER_ROW;
      int ce  = (c % CH_PER_ROW) * EL_PER_CH;
      async_copy16(A + (size_t)(m0 + row) * Ksz + k0 + ce, &As[row][ce]);
    }

    // ---- stage B tile transposed: 32x64 f32 -> Bts[n][k] bf16 ----
#pragma unroll
    for (int pass = 0; pass < 2; ++pass) {
      int li = (pass * 256 + tid) * 4;
      int kr = li >> 6;
      int nc = li & 63;
      const float* src = Wt + (size_t)(k0 + kr) * Nsz + n0 + nc;
      float4 v = *reinterpret_cast<const float4*>(src);
      Bts[nc + 0][kr] = (__bf16)v.x;
      Bts[nc + 1][kr] = (__bf16)v.y;
      Bts[nc + 2][kr] = (__bf16)v.z;
      Bts[nc + 3][kr] = (__bf16)v.w;
    }
    // prefetch next K-slab of the weight matrix into GL2
    if (kt + 1 < Ksz / 32)
      __builtin_prefetch(Wt + (size_t)(k0 + 32 + (tid >> 6)) * Nsz + n0, 0, 1);

    wait_async0();
    __syncthreads();

    v16bf a;
    if constexpr (sizeof(AT) == 4)
      a = load_frag_a_f32((const float*)&As[wave * 16][0], 32);
    else
      a = load_frag_a((const __bf16*)&As[wave * 16][0], 32);

#pragma unroll
    for (int nt = 0; nt < 4; ++nt) {
      v16bf b = load_frag_b(&Bts[nt * 16][0], 32);
      acc[nt] = wmma_bf16(a, b, acc[nt]);
    }
    __syncthreads();
  }

  // ---- epilogue: C layout = col (lane&15), rows r + 8*(lane>=16) ----
  const int lane = tid & 31;
  const int colc = lane & 15;
  const int row0 = (lane >> 4) * 8;
#pragma unroll
  for (int nt = 0; nt < 4; ++nt) {
    const int n  = n0 + nt * 16 + colc;
    const float bv = bias[n];
#pragma unroll
    for (int r = 0; r < 8; ++r) {
      const int m   = m0 + wave * 16 + row0 + r;
      const float v = acc[nt][r] + bv;
      if (MODE == 2) {
        outf[(size_t)m * Nsz + n] = v;
      } else {
        const int b  = m >> 11;      // m / L
        const int l  = m & (L - 1);
        const int h  = n >> 6;       // n / Dh
        const int dh = n & (Dh - 1);
        size_t idx;
        if (MODE == 0)
          idx = ((size_t)(b * H + h) * L + l) * Dh + dh;
        else
          idx = ((size_t)(b * H + h) * Dh + dh) * (size_t)L + l;
        outbf[idx] = (__bf16)v;
      }
    }
  }
}

// ---------------------------------------------------------------------------
// Kernel 2/5: masked scaled scores = (Qp Kp^T / 8) * sigmoid(W)
// Qp,Kp bf16 [bh][l][64].  Per wave: 16 q-rows x 128 k-cols, K=64 (2 WMMAs/tile)
// Grid: (L/1024, L/16, B*H), 256 threads.  Fragments straight from global.
// ---------------------------------------------------------------------------
__global__ __launch_bounds__(256)
void scores_kernel(const __bf16* __restrict__ Qp, const __bf16* __restrict__ Kp,
                   const float* __restrict__ W, float* __restrict__ attnW) {
  const int tid  = threadIdx.x;
  const int wave = tid >> 5;
  const int lane = tid & 31;
  const int bh   = blockIdx.z;
  const int q0   = blockIdx.y * 16;
  const int n0   = blockIdx.x * 1024 + wave * 128;
  const int b    = bh >> 4;

  const __bf16* qbase = Qp + ((size_t)bh * L + q0) * Dh;
  v16bf a0 = load_frag_a(qbase, Dh);       // K = 0..31  of Dh
  v16bf a1 = load_frag_a(qbase + 32, Dh);  // K = 32..63

  const int colc = lane & 15;
  const int row0 = (lane >> 4) * 8;

#pragma unroll
  for (int c = 0; c < 8; ++c) {
    const int n = n0 + c * 16;
    // B = Kp^T: lane's column = key row (n+colc); contiguous over Dh.
    const __bf16* kbase = Kp + ((size_t)bh * L + n) * Dh;
    v8f acc = {};
    v16bf b0 = load_frag_b(kbase, Dh);
    v16bf b1 = load_frag_b(kbase + 32, Dh);
    acc = wmma_bf16(a0, b0, acc);
    acc = wmma_bf16(a1, b1, acc);

    const int kg = n + colc;
#pragma unroll
    for (int r = 0; r < 8; ++r) {
      const int q   = q0 + row0 + r;
      const float s = acc[r] * 0.125f;  // 1/sqrt(64)
      const float w = W[((size_t)b * L + q) * L + kg];
      const float gate = 1.0f / (1.0f + __expf(-w));
      attnW[((size_t)bh * L + q) * L + kg] = s * gate;
    }
  }
}

// ---------------------------------------------------------------------------
// Kernel 3/5: in-place row softmax over 2048 cols.  One block per row.
// ---------------------------------------------------------------------------
__global__ __launch_bounds__(256)
void softmax_kernel(float* __restrict__ attnW) {
  __shared__ float red[256];
  const int tid = threadIdx.x;
  const size_t base = (size_t)blockIdx.x * L;

  float v[8];
#pragma unroll
  for (int j = 0; j < 8; ++j) v[j] = attnW[base + tid + j * 256];

  float mx = v[0];
#pragma unroll
  for (int j = 1; j < 8; ++j) mx = fmaxf(mx, v[j]);
  red[tid] = mx;
  __syncthreads();
  for (int s = 128; s > 0; s >>= 1) {
    if (tid < s) red[tid] = fmaxf(red[tid], red[tid + s]);
    __syncthreads();
  }
  mx = red[0];
  __syncthreads();

  float sum = 0.0f;
#pragma unroll
  for (int j = 0; j < 8; ++j) {
    v[j] = __expf(v[j] - mx);
    sum += v[j];
  }
  red[tid] = sum;
  __syncthreads();
  for (int s = 128; s > 0; s >>= 1) {
    if (tid < s) red[tid] += red[tid + s];
    __syncthreads();
  }
  const float inv = 1.0f / red[0];
#pragma unroll
  for (int j = 0; j < 8; ++j) attnW[base + tid + j * 256] = v[j] * inv;
}

// ---------------------------------------------------------------------------
// Kernel 4/5: attn_out = P @ V.  P f32 [bh][q][k] (converted to bf16 on the
// fly), Vt bf16 [bh][dh][L] (transposed so B-fragments are contiguous).
// Per wave: 16 q-rows x 64 dh, K-loop over 2048 in steps of 32.
// Output bf16 merged-heads [b][q][D].
// ---------------------------------------------------------------------------
__global__ __launch_bounds__(256)
void attn_v_kernel(const float* __restrict__ P, const __bf16* __restrict__ Vt,
                   __bf16* __restrict__ attnOut) {
  const int tid  = threadIdx.x;
  const int wave = tid >> 5;
  const int lane = tid & 31;
  const int bh   = blockIdx.y;
  const int q0   = blockIdx.x * 128 + wave * 16;
  const int b    = bh >> 4;
  const int h    = bh & 15;

  v8f zero = {};
  v8f acc[4];
#pragma unroll
  for (int i = 0; i < 4; ++i) acc[i] = zero;

  const float*  pbase = P + ((size_t)bh * L + q0) * L;
  const __bf16* vbase = Vt + (size_t)bh * Dh * L;

  for (int kt = 0; kt < L / 32; ++kt) {
    const int k0 = kt * 32;
    v16bf a = load_frag_a_f32(pbase + k0, L);
#pragma unroll
    for (int nt = 0; nt < 4; ++nt) {
      v16bf bb = load_frag_b(vbase + (size_t)(nt * 16) * L + k0, L);
      acc[nt] = wmma_bf16(a, bb, acc[nt]);
    }
  }

  const int colc = lane & 15;
  const int row0 = (lane >> 4) * 8;
#pragma unroll
  for (int nt = 0; nt < 4; ++nt) {
    const int col = h * Dh + nt * 16 + colc;
#pragma unroll
    for (int r = 0; r < 8; ++r) {
      const int q = q0 + row0 + r;
      attnOut[((size_t)b * L + q) * D + col] = (__bf16)acc[nt][r];
    }
  }
}

// ---------------------------------------------------------------------------
// Host launcher
// ---------------------------------------------------------------------------
extern "C" void kernel_launch(void* const* d_in, const int* in_sizes, int n_in,
                              void* d_out, int out_size, void* d_ws,
                              size_t ws_size, hipStream_t stream) {
  const float* Q  = (const float*)d_in[0];
  const float* K  = (const float*)d_in[1];
  const float* W  = (const float*)d_in[2];
  const float* Wq = (const float*)d_in[3];
  const float* bq = (const float*)d_in[4];
  const float* Wk = (const float*)d_in[5];
  const float* bk = (const float*)d_in[6];
  const float* Wv = (const float*)d_in[7];
  const float* bv = (const float*)d_in[8];
  const float* Wo = (const float*)d_in[9];
  const float* bo = (const float*)d_in[10];

  float* out   = (float*)d_out;
  float* attnW = out + (size_t)BSZ * L * D;  // attn_weights slot of d_out

  const size_t projElems = (size_t)BSZ * H * L * Dh;  // 4,194,304
  __bf16* Qp = (__bf16*)d_ws;
  __bf16* Kp = Qp + projElems;
  __bf16* Vt = Kp + projElems;
  __bf16* AO = Vt + projElems;  // merged-head attention output (bf16)

  dim3 blk(256);
  dim3 gProj(D / 64, (BSZ * L) / 128);  // (16, 32)

  // Q/K/V projections (bf16, head-split; V transposed)
  gemm_bias_kernel<float, 0><<<gProj, blk, 0, stream>>>(Q, Wq, bq, Qp, nullptr);
  gemm_bias_kernel<float, 0><<<gProj, blk, 0, stream>>>(K, Wk, bk, Kp, nullptr);
  gemm_bias_kernel<float, 1><<<gProj, blk, 0, stream>>>(K, Wv, bv, Vt, nullptr);

  // masked scaled scores -> attn_weights buffer
  dim3 gS(L / 1024, L / 16, BSZ * H);  // (2, 128, 32)
  scores_kernel<<<gS, blk, 0, stream>>>(Qp, Kp, W, attnW);

  // in-place softmax (one block per row)
  softmax_kernel<<<dim3(BSZ * H * L), blk, 0, stream>>>(attnW);

  // attn_out = P @ V (merged heads, bf16)
  attn_v_kernel<<<dim3(L / 128, BSZ * H), blk, 0, stream>>>(attnW, Vt, AO);

  // output projection (f32 out + bias)
  gemm_bias_kernel<__bf16, 2><<<gProj, blk, 0, stream>>>(AO, Wo, bo, nullptr, out);
}
